// MGKN_21852793602344
// MI455X (gfx1250) — compile-verified, matrix operand order checked by reading, not run on hardware
//
#include <hip/hip_runtime.h>

typedef __attribute__((ext_vector_type(16))) __bf16 v16bf;
typedef __attribute__((ext_vector_type(8)))  float  v8f;
typedef unsigned short ushort_t;

union BFr { v16bf v; uint4 q[2]; unsigned u[8]; };

__device__ __forceinline__ unsigned short f2bf(float f) {
    union { __bf16 b; unsigned short s; } u; u.b = (__bf16)f; return u.s;
}
__device__ __forceinline__ unsigned packbf(float a, float b) {
    return (unsigned)f2bf(a) | ((unsigned)f2bf(b) << 16);
}
// packed bf16 multiply with src1 half-broadcast via VOP3P OPSEL:
//  lo-broadcast: d.lo = a.lo*b.lo, d.hi = a.hi*b.lo
__device__ __forceinline__ unsigned pkmul_bf16_lo(unsigned a, unsigned b) {
    unsigned d;
    asm("v_pk_mul_bf16 %0, %1, %2 op_sel:[0,0] op_sel_hi:[1,0]"
        : "=v"(d) : "v"(a), "v"(b));
    return d;
}
//  hi-broadcast: d.lo = a.lo*b.hi, d.hi = a.hi*b.hi
__device__ __forceinline__ unsigned pkmul_bf16_hi(unsigned a, unsigned b) {
    unsigned d;
    asm("v_pk_mul_bf16 %0, %1, %2 op_sel:[0,1] op_sel_hi:[1,1]"
        : "=v"(d) : "v"(a), "v"(b));
    return d;
}
__device__ __forceinline__ v8f wmma_bf16(const BFr& a, const BFr& b, v8f c) {
    return __builtin_amdgcn_wmma_f32_16x16x32_bf16(false, a.v, false, b.v,
                                                   (short)0, c, false, false);
}
__device__ __forceinline__ void atomAddF(float* p, float v) {
    __hip_atomic_fetch_add(p, v, __ATOMIC_RELAXED, __HIP_MEMORY_SCOPE_AGENT);
}
// Async Global->LDS DMA (16B per lane), tracked by ASYNCcnt (ISA 08_async_tensor §4)
__device__ __forceinline__ void async_copy16(unsigned lds_off, const void* g) {
    asm volatile("global_load_async_to_lds_b128 %0, %1, off"
                 :: "v"(lds_off), "v"(g) : "memory");
}
__device__ __forceinline__ void wait_async0() {
    asm volatile("s_wait_asynccnt 0" ::: "memory");
}

// ---------------------------------------------------------------- prep kernels
__global__ void k_prep_w2t(const float* __restrict__ W2, ushort_t* __restrict__ W2T,
                           int kw, int k2p) {
    int id = blockIdx.x * blockDim.x + threadIdx.x;
    if (id >= kw * k2p) return;
    int n = id / k2p, k = id % k2p;
    W2T[id] = (k < kw) ? f2bf(W2[k * kw + n]) : (ushort_t)0;
}

__global__ void k_prep_w3t(const float* __restrict__ W3, const float* __restrict__ b3,
                           ushort_t* __restrict__ W3T, int kw, int kp) {
    int id = blockIdx.x * blockDim.x + threadIdx.x;
    if (id >= kp * 1024) return;
    int k = id >> 10, r = id & 1023, o = r >> 5, i = r & 31;
    float v = (k < kw) ? W3[k * 1024 + i * 32 + o] : (k == kw ? b3[i * 32 + o] : 0.0f);
    W3T[id] = f2bf(v);
}

__global__ void k_cnt(const int* __restrict__ tgt, float* __restrict__ cnt, int E) {
    int id = blockIdx.x * blockDim.x + threadIdx.x;
    if (id < E) atomAddF(&cnt[tgt[id]], 1.0f);
}

__global__ void k_cvt(const float* __restrict__ x, ushort_t* __restrict__ out, int n) {
    int id = blockIdx.x * blockDim.x + threadIdx.x;
    if (id < n) out[id] = f2bf(x[id]);
}

__global__ void k_fc1(const float* __restrict__ x0, const float* __restrict__ W,
                      const float* __restrict__ b, float* __restrict__ out) {
    int id = blockIdx.x * blockDim.x + threadIdx.x;   // 4096*32
    int v = id >> 5, o = id & 31;
    float s = b[o];
#pragma unroll
    for (int j = 0; j < 6; j++) s += x0[v * 6 + j] * W[j * 32 + o];
    out[id] = s;
}

__global__ void k_down(const float* __restrict__ in, float* __restrict__ out, int n_out) {
    int id = blockIdx.x * blockDim.x + threadIdx.x;   // n_out*32
    if (id >= n_out * 32) return;
    int v = id >> 5, o = id & 31;
    out[id] = 0.5f * (in[(2 * v) * 32 + o] + in[(2 * v + 1) * 32 + o]);
}

// ---------------------------------------------------------------- edge MLP (WMMA)
// h2 = relu(relu(ea@W1+b1)@W2+b2), bf16 out [E,KP], plus h[e][KW]=1, pad zeros
template <int KW>
__global__ __launch_bounds__(256) void k_edge_mlp(
    const float* __restrict__ ea, const float* __restrict__ W1,
    const float* __restrict__ b1, const float* __restrict__ b2,
    const ushort_t* __restrict__ W2T, ushort_t* __restrict__ h, int KP) {
    constexpr int K2P = (KW < 32) ? 32 : KW;
    constexpr int KCH = K2P / 32;
    constexpr int NCH = KW / 16;
    extern __shared__ char smem[];
    float*    W1s  = (float*)smem;                       // 6*KW
    float*    b1s  = W1s + 6 * KW;                       // KW
    float*    b2s  = b1s + KW;                           // KW
    ushort_t* W2Ts = (ushort_t*)(b2s + KW);              // KW*K2P
    float*    msgs = (float*)(W2Ts + (size_t)KW * K2P);  // 8*256

    int tid = threadIdx.x;
    for (int i = tid; i < 6 * KW; i += 256) W1s[i] = W1[i];
    for (int i = tid; i < KW; i += 256) { b1s[i] = b1[i]; b2s[i] = b2[i]; }
    {   // async DMA the bf16 W2^T panel into LDS
        unsigned base = (unsigned)((char*)W2Ts - smem);
        const char* g = (const char*)W2T;
        for (int i = tid; i < KW * K2P / 8; i += 256)
            async_copy16(base + i * 16, g + (size_t)i * 16);
        wait_async0();
    }
    __syncthreads();

    int wave = tid >> 5, lane = tid & 31;
    int row = lane & 15, half = lane >> 4;
    long ebase = (long)(blockIdx.x * 8 + wave) * 16;
    long edge  = ebase + row;

    float ear[6];
#pragma unroll
    for (int j = 0; j < 6; j++) ear[j] = ea[edge * 6 + j];

    // layer 1 (K=6, VALU) directly into A-layout bf16 packs
    unsigned apack[KCH][8];
#pragma unroll
    for (int c = 0; c < KCH; c++) {
#pragma unroll
        for (int p = 0; p < 8; p++) {
            int kl = ((p < 4) ? 2 * p : 2 * p + 8) + half * 8;
            int col0 = 32 * c + kl;
            float t1a = 0.0f, t1b = 0.0f;
            if (col0 < KW) {
                t1a = b1s[col0]; t1b = b1s[col0 + 1];
#pragma unroll
                for (int j = 0; j < 6; j++) {
                    t1a += ear[j] * W1s[j * KW + col0];
                    t1b += ear[j] * W1s[j * KW + col0 + 1];
                }
                t1a = fmaxf(t1a, 0.0f); t1b = fmaxf(t1b, 0.0f);
            }
            apack[c][p] = packbf(t1a, t1b);
        }
    }

    float* mymsg = msgs + wave * 256;
    for (int nt = 0; nt < NCH; nt++) {
        v8f acc = {};
        int ncol = nt * 16 + row;
        const ushort_t* brow = W2Ts + (size_t)ncol * K2P + half * 8;
        uint4 bb[2][2];                 // double-buffered B tiles
        bb[0][0] = *(const uint4*)(brow);
        bb[0][1] = *(const uint4*)(brow + 16);
#pragma unroll
        for (int c = 0; c < KCH; c++) {
            if (c + 1 < KCH) {
                bb[(c + 1) & 1][0] = *(const uint4*)(brow + 32 * (c + 1));
                bb[(c + 1) & 1][1] = *(const uint4*)(brow + 32 * (c + 1) + 16);
            }
            BFr a, b;
#pragma unroll
            for (int j = 0; j < 8; j++) a.u[j] = apack[c][j];
            b.q[0] = bb[c & 1][0];
            b.q[1] = bb[c & 1][1];
            acc = wmma_bf16(a, b, acc);
        }
        float bc = b2s[nt * 16 + row];
#pragma unroll
        for (int j = 0; j < 8; j++)
            mymsg[(j + 8 * half) * 16 + row] = fmaxf(acc[j] + bc, 0.0f);
        // LDS-bounce to coalesced bf16-pair stores of h tile
#pragma unroll
        for (int t = 0; t < 4; t++) {
            int idx = lane + 32 * t;        // 0..127
            int r2 = idx >> 3, cp = idx & 7;
            unsigned pv = packbf(mymsg[r2 * 16 + 2 * cp], mymsg[r2 * 16 + 2 * cp + 1]);
            long er = ebase + r2;
            *(unsigned*)(h + er * KP + nt * 16 + 2 * cp) = pv;
        }
    }
    // bias row (=1.0) + zero pad
    if (lane < 16) {
        ushort_t* hp = h + (ebase + lane) * KP;
        hp[KW] = 0x3f80;
        for (int c = KW + 1; c < KP; c++) hp[c] = 0;
    }
}

// ---------------------------------------------------------------- message (WMMA)
// msg[e,:] = x_src[e] @ reshape(h[e]@W3+b3, 32,32), atomically scattered into agg
// Each wave owns TWO 16-edge tiles so every LDS B fetch feeds 4 WMMAs.
__global__ __launch_bounds__(256) void k_message(
    const ushort_t* __restrict__ xbf, const int* __restrict__ src,
    const int* __restrict__ tgt, const ushort_t* __restrict__ h,
    const ushort_t* __restrict__ W3T, float* __restrict__ agg, int KP) {
    extern __shared__ char smem[];
    ushort_t* W3s  = (ushort_t*)smem;            // 32*1024 bf16 (64KB)
    float*    msgs = (float*)(W3s + 32 * 1024);  // 8*1024 f32 (32KB)
    int*      tgts = (int*)(msgs + 8 * 1024);    // 8*32

    int tid = threadIdx.x, wave = tid >> 5, lane = tid & 31;
    int row = lane & 15, half = lane >> 4;
    long ebase = (long)(blockIdx.x * 8 + wave) * 32;
    long edge0 = ebase + row;
    long edge1 = ebase + 16 + row;
    int s0 = src[edge0];
    int s1 = src[edge1];

    BFr xa0, xa1;  // x_src rows in A layout (cached across whole K loop)
    xa0.q[0] = *(const uint4*)(xbf + (long)s0 * 32 + half * 8);
    xa0.q[1] = *(const uint4*)(xbf + (long)s0 * 32 + half * 8 + 16);
    xa1.q[0] = *(const uint4*)(xbf + (long)s1 * 32 + half * 8);
    xa1.q[1] = *(const uint4*)(xbf + (long)s1 * 32 + half * 8 + 16);

    v8f acc00 = {}, acc01 = {}, acc10 = {}, acc11 = {};
    const ushort_t* Bb0 = W3s + ((0 * 16 + row) * 32 + half * 8);
    const ushort_t* Bb1 = W3s + ((1 * 16 + row) * 32 + half * 8);
    unsigned w3base = (unsigned)((char*)W3s - smem);   // dynamic LDS starts at 0

    int nch = KP >> 5;
    for (int ch = 0; ch < nch; ch++) {
        __syncthreads();
        {   // async DMA 32 k-rows of W3T (64KB) into LDS
            const char* g = (const char*)(W3T + (size_t)ch * 32 * 1024);
            for (int i = tid; i < 4096; i += 256)
                async_copy16(w3base + i * 16, g + (size_t)i * 16);
            wait_async0();
        }
        __syncthreads();
        uint4 hp0[4], hp1[4];
        {   // each tile's h[e, ch*32 .. +31]
            const uint4* hg0 = (const uint4*)(h + edge0 * (long)KP + ch * 32);
            const uint4* hg1 = (const uint4*)(h + edge1 * (long)KP + ch * 32);
            hp0[0] = hg0[0]; hp0[1] = hg0[1]; hp0[2] = hg0[2]; hp0[3] = hg0[3];
            hp1[0] = hg1[0]; hp1[1] = hg1[1]; hp1[2] = hg1[2]; hp1[3] = hg1[3];
        }
        uint4 bb[2][4];                 // double-buffered B tiles
        bb[0][0] = *(const uint4*)(Bb0);
        bb[0][1] = *(const uint4*)(Bb0 + 16);
        bb[0][2] = *(const uint4*)(Bb1);
        bb[0][3] = *(const uint4*)(Bb1 + 16);
#pragma unroll
        for (int kk = 0; kk < 32; kk++) {
            if (kk + 1 < 32) {          // prefetch next k's B tiles
                int nb = (kk + 1) & 1;
                bb[nb][0] = *(const uint4*)(Bb0 + (kk + 1) * 1024);
                bb[nb][1] = *(const uint4*)(Bb0 + (kk + 1) * 1024 + 16);
                bb[nb][2] = *(const uint4*)(Bb1 + (kk + 1) * 1024);
                bb[nb][3] = *(const uint4*)(Bb1 + (kk + 1) * 1024 + 16);
            }
            unsigned hw0 = ((const unsigned*)hp0)[kk >> 1];
            unsigned hw1 = ((const unsigned*)hp1)[kk >> 1];
            BFr a0, a1;
            if (kk & 1) {
#pragma unroll
                for (int j = 0; j < 8; j++) {
                    a0.u[j] = pkmul_bf16_hi(xa0.u[j], hw0);
                    a1.u[j] = pkmul_bf16_hi(xa1.u[j], hw1);
                }
            } else {
#pragma unroll
                for (int j = 0; j < 8; j++) {
                    a0.u[j] = pkmul_bf16_lo(xa0.u[j], hw0);
                    a1.u[j] = pkmul_bf16_lo(xa1.u[j], hw1);
                }
            }
            BFr b0, b1;
            b0.q[0] = bb[kk & 1][0]; b0.q[1] = bb[kk & 1][1];
            b1.q[0] = bb[kk & 1][2]; b1.q[1] = bb[kk & 1][3];
            acc00 = wmma_bf16(a0, b0, acc00);
            acc01 = wmma_bf16(a0, b1, acc01);
            acc10 = wmma_bf16(a1, b0, acc10);
            acc11 = wmma_bf16(a1, b1, acc11);
        }
    }
    // stash msg tiles [32 rows][32 cols] and scatter with fp32 atomics
    float* mymsg = msgs + wave * 1024;
#pragma unroll
    for (int j = 0; j < 8; j++) {
        mymsg[(j + 8 * half) * 32 + row]             = acc00[j];
        mymsg[(j + 8 * half) * 32 + 16 + row]        = acc01[j];
        mymsg[(16 + j + 8 * half) * 32 + row]        = acc10[j];
        mymsg[(16 + j + 8 * half) * 32 + 16 + row]   = acc11[j];
    }
    tgts[wave * 32 + lane] = tgt[ebase + lane];
#pragma unroll
    for (int t = 0; t < 32; t++) {
        int idx = lane + 32 * t;  // 0..1023
        int r2 = idx >> 5, c2 = idx & 31;
        atomAddF(&agg[(long)tgts[wave * 32 + r2] * 32 + c2], mymsg[idx]);
    }
}

// ---------------------------------------------------------------- finalize
// out = relu(prev + agg/max(cnt,1) + x_in@root + bias)
__global__ void k_finalize(const float* __restrict__ agg, const float* __restrict__ cnt,
                           const float* __restrict__ xin, const float* __restrict__ root,
                           const float* __restrict__ bias, const float* __restrict__ prev,
                           int shift, float* __restrict__ out, int n) {
    int id = blockIdx.x * blockDim.x + threadIdx.x;
    if (id >= n * 32) return;
    int v = id >> 5, o = id & 31;
    float a = agg[id] / fmaxf(cnt[v], 1.0f);
    float r = bias[o];
    const float* xr = xin + (long)v * 32;
#pragma unroll
    for (int i = 0; i < 32; i++) r += xr[i] * root[i * 32 + o];
    float p = prev[(long)(v >> shift) * 32 + o];
    out[id] = fmaxf(p + a + r, 0.0f);
}

__global__ void k_fc2(const float* __restrict__ x, const float* __restrict__ W,
                      const float* __restrict__ b, float* __restrict__ out) {
    int id = blockIdx.x * blockDim.x + threadIdx.x;  // 4096*256
    int v = id >> 8, j = id & 255;
    float s = b[j];
#pragma unroll
    for (int i = 0; i < 32; i++) s += x[v * 32 + i] * W[i * 256 + j];
    out[id] = fmaxf(s, 0.0f);
}

__global__ void k_fc3(const float* __restrict__ t2, const float* __restrict__ W,
                      const float* __restrict__ b, float* __restrict__ out) {
    int v = blockIdx.x * blockDim.x + threadIdx.x;  // 4096
    float s = b[0];
    for (int j = 0; j < 256; j++) s += t2[v * 256 + j] * W[j];
    out[v] = s;
}

// ---------------------------------------------------------------- host
struct Conv {
    int kw, k2p, kp;
    const float *W1, *W2, *W3, *b1, *b2, *b3, *bias, *root;
    ushort_t *w2t, *w3t;
};

extern "C" void kernel_launch(void* const* d_in, const int* in_sizes, int n_in,
                              void* d_out, int out_size, void* d_ws, size_t ws_size,
                              hipStream_t stream) {
    (void)n_in; (void)out_size; (void)ws_size;
    char* ws = (char*)d_ws;
    size_t off = 0;
    auto carve = [&](size_t bytes) -> char* {
        char* p = ws + off;
        off += (bytes + 255) & ~(size_t)255;
        return p;
    };

    float* XA   = (float*)carve(4096 * 32 * 4);
    float* XB   = (float*)carve(4096 * 32 * 4);
    float* xc0  = (float*)carve(4096 * 32 * 4);
    float* xc1  = (float*)carve(2048 * 32 * 4);
    float* xc2  = (float*)carve(1024 * 32 * 4);
    float* xc3  = (float*)carve(512 * 32 * 4);
    float* xc4  = (float*)carve(256 * 32 * 4);
    float* phi1 = (float*)carve(2048 * 32 * 4);
    float* phi2 = (float*)carve(1024 * 32 * 4);
    float* phi3 = (float*)carve(512 * 32 * 4);
    float* phi4 = (float*)carve(256 * 32 * 4);
    ushort_t* xbf = (ushort_t*)carve(4096 * 32 * 2);
    float* agg  = (float*)carve(4096 * 32 * 4);
    float* cnt  = (float*)carve(4096 * 4);
    ushort_t* hbuf = (ushort_t*)carve((size_t)65536 * 288 * 2);
    float* t2   = (float*)carve((size_t)4096 * 256 * 4);

    auto mkconv = [&](int base, int kw) -> Conv {
        Conv c;
        c.kw = kw;
        c.k2p = kw < 32 ? 32 : kw;
        c.kp = ((kw + 1 + 31) / 32) * 32;
        c.W1 = (const float*)d_in[base + 0]; c.W2 = (const float*)d_in[base + 1];
        c.W3 = (const float*)d_in[base + 2]; c.b1 = (const float*)d_in[base + 3];
        c.b2 = (const float*)d_in[base + 4]; c.b3 = (const float*)d_in[base + 5];
        c.bias = (const float*)d_in[base + 6]; c.root = (const float*)d_in[base + 7];
        c.w2t = (ushort_t*)carve((size_t)kw * c.k2p * 2);
        c.w3t = (ushort_t*)carve((size_t)c.kp * 1024 * 2);
        return c;
    };
    // pytree leaf order: conv_coarse, convs[0..4] (inner dicts key-sorted)
    Conv coarse = mkconv(11, 16);
    Conv convs[5] = { mkconv(19, 256), mkconv(27, 128), mkconv(35, 64),
                      mkconv(43, 32),  mkconv(51, 16) };
    const float* fc1W = (const float*)d_in[59];
    const float* fc1b = (const float*)d_in[60];
    const float* fc2W = (const float*)d_in[61];
    const float* fc2b = (const float*)d_in[62];
    const float* fc3W = (const float*)d_in[63];
    const float* fc3b = (const float*)d_in[64];
    const float* x0 = (const float*)d_in[0];

    // per-launch weight prep (bf16 + layout transforms)
    auto prep = [&](const Conv& c) {
        k_prep_w2t<<<(c.kw * c.k2p + 255) / 256, 256, 0, stream>>>(c.W2, c.w2t, c.kw, c.k2p);
        k_prep_w3t<<<(c.kp * 1024 + 255) / 256, 256, 0, stream>>>(c.W3, c.b3, c.w3t, c.kw, c.kp);
    };
    prep(coarse);
    for (int i = 0; i < 5; i++) prep(convs[i]);

    auto launch_mlp = [&](const Conv& c, const float* ea, int E) {
        size_t lds = (size_t)6 * c.kw * 4 + (size_t)c.kw * 8 +
                     (size_t)c.kw * c.k2p * 2 + 8 * 256 * 4;
        dim3 g(E / 128), b(256);
        switch (c.kw) {
            case 16:  k_edge_mlp<16><<<g, b, lds, stream>>>(ea, c.W1, c.b1, c.b2, c.w2t, hbuf, c.kp); break;
            case 32:  k_edge_mlp<32><<<g, b, lds, stream>>>(ea, c.W1, c.b1, c.b2, c.w2t, hbuf, c.kp); break;
            case 64:  k_edge_mlp<64><<<g, b, lds, stream>>>(ea, c.W1, c.b1, c.b2, c.w2t, hbuf, c.kp); break;
            case 128: k_edge_mlp<128><<<g, b, lds, stream>>>(ea, c.W1, c.b1, c.b2, c.w2t, hbuf, c.kp); break;
            default:  k_edge_mlp<256><<<g, b, lds, stream>>>(ea, c.W1, c.b1, c.b2, c.w2t, hbuf, c.kp); break;
        }
    };

    auto conv_step = [&](const Conv& c, int lvl, const float* xin, int n,
                         const float* prev, int shift, float* out) {
        const int* ei = (const int*)d_in[1 + 2 * lvl];
        const float* ea = (const float*)d_in[2 + 2 * lvl];
        int E = in_sizes[1 + 2 * lvl] / 2;
        const int* srcI = ei;
        const int* tgtI = ei + E;
        hipMemsetAsync(agg, 0, (size_t)n * 32 * 4, stream);
        hipMemsetAsync(cnt, 0, (size_t)n * 4, stream);
        k_cnt<<<E / 256, 256, 0, stream>>>(tgtI, cnt, E);
        k_cvt<<<n * 32 / 256, 256, 0, stream>>>(xin, xbf, n * 32);
        launch_mlp(c, ea, E);
        k_message<<<E / 256, 256, 32 * 1024 * 2 + 8 * 1024 * 4 + 8 * 32 * 4, stream>>>(
            xbf, srcI, tgtI, hbuf, c.w3t, agg, c.kp);
        k_finalize<<<n * 32 / 256, 256, 0, stream>>>(agg, cnt, xin, c.root, c.bias,
                                                     prev, shift, out, n);
    };

    k_fc1<<<4096 * 32 / 256, 256, 0, stream>>>(x0, fc1W, fc1b, XA);
    float* xcur = XA;
    float* xalt = XB;
    for (int d = 0; d < 2; d++) {
        k_down<<<2048 * 32 / 256, 256, 0, stream>>>(xcur, phi1, 2048);
        k_down<<<1024 * 32 / 256, 256, 0, stream>>>(phi1, phi2, 1024);
        k_down<<<512 * 32 / 256, 256, 0, stream>>>(phi2, phi3, 512);
        k_down<<<256 * 32 / 256, 256, 0, stream>>>(phi3, phi4, 256);
        conv_step(coarse,   4, phi4, 256,  phi4, 0, xc4);
        conv_step(convs[4], 4, phi3, 512,  xc4,  1, xc3);
        conv_step(convs[3], 3, phi2, 1024, xc3,  1, xc2);
        conv_step(convs[2], 2, phi1, 2048, xc2,  1, xc1);
        conv_step(convs[1], 1, xcur, 4096, xc1,  1, xc0);
        conv_step(convs[0], 0, xcur, 4096, xc0,  0, xalt);
        float* t = xcur; xcur = xalt; xalt = t;
    }
    k_fc2<<<4096 * 256 / 256, 256, 0, stream>>>(xcur, fc2W, fc2b, t2);
    k_fc3<<<4096 / 256, 256, 0, stream>>>(t2, fc3W, fc3b, (float*)d_out);
}
